// AttentionBlock_90881507983922
// MI455X (gfx1250) — matrix operand order repaired and checked by reading, not compiled, not run on hardware
//
#include <hip/hip_runtime.h>
#include <hip/hip_bf16.h>

typedef unsigned short u16;
typedef unsigned int   u32;
typedef __attribute__((ext_vector_type(16))) __bf16 v16bf;
typedef __attribute__((ext_vector_type(8)))  float  v8f;

union FragBF { v16bf v; uint4 u[2]; };

__device__ inline u16 f2bf(float f) {
    u32 x = __float_as_uint(f);
    u32 r = (x + 0x7fffu + ((x >> 16) & 1u)) >> 16;   // round-to-nearest-even
    return (u16)r;
}
__device__ inline uint2 pack4bf(float4 f) {
    u32 a = (u32)f2bf(f.x) | ((u32)f2bf(f.y) << 16);
    u32 b = (u32)f2bf(f.z) | ((u32)f2bf(f.w) << 16);
    return make_uint2(a, b);
}
__device__ inline v8f vzero8() {
    v8f z = {0.f, 0.f, 0.f, 0.f, 0.f, 0.f, 0.f, 0.f};
    return z;
}
__device__ inline v8f bfwmma(v16bf a, v16bf b, v8f c) {
    return __builtin_amdgcn_wmma_f32_16x16x32_bf16(false, a, false, b, (short)0, c, false, false);
}
// A-matrix fragment (16x32 bf16): lane m = lane%16, elements K = g*16 + 0..15.
__device__ inline v16bf loadFragA(const u16* p) {
    FragBF f;
    f.u[0] = *(const uint4*)(p);
    f.u[1] = *(const uint4*)(p + 8);
    return f.v;
}
// B-matrix fragment (32x16 bf16) from [N][K]-layout rows: lane n = lane%16,
// elements K = g*16 + 0..15.
__device__ inline v16bf loadFragB(const u16* p) {
    FragBF f;
    f.u[0] = *(const uint4*)(p);
    f.u[1] = *(const uint4*)(p + 8);
    return f.v;
}

// --- CDNA5 async global->LDS DMA (ASYNCcnt-tracked), per-lane 16B ------------
// LDS operand is the workgroup-relative LDS byte offset (generic-pointer low
// 32 bits).
__device__ inline void async_ld_b128(const void* lds_ptr, const void* gaddr) {
    u32 lds_off = (u32)(size_t)lds_ptr;
    asm volatile("global_load_async_to_lds_b128 %0, %1, off"
                 :: "v"(lds_off), "v"(gaddr) : "memory");
}
__device__ inline void wait_async0() {
    asm volatile("s_wait_asynccnt 0x0" ::: "memory");
}

// ---------------------------------------------------------------------------
// Kernel 1: fused QKV projection.  C[4096x1024] = act[4096x1024] x W^T, +bias.
// Output bf16 in [B][H][S][HS] layout for attention.  (fp32 source -> must
// convert during staging, so this kernel keeps the manual VGPR path.)
// ---------------------------------------------------------------------------
__global__ __launch_bounds__(256) void qkv_proj_kernel(
    const float* __restrict__ q_in, const float* __restrict__ k_in, const float* __restrict__ v_in,
    const float* __restrict__ Wq, const float* __restrict__ Wk, const float* __restrict__ Wv,
    const float* __restrict__ bq, const float* __restrict__ bk, const float* __restrict__ bv,
    u16* __restrict__ qb, u16* __restrict__ kb, u16* __restrict__ vb)
{
    __shared__ u16 As[128 * 40];
    __shared__ u16 Bs[128 * 40];

    const int which = blockIdx.z;
    const float* act  = which == 0 ? q_in : (which == 1 ? k_in : v_in);
    const float* W    = which == 0 ? Wq   : (which == 1 ? Wk   : Wv);
    const float* bias = which == 0 ? bq   : (which == 1 ? bk   : bv);
    u16* out          = which == 0 ? qb   : (which == 1 ? kb   : vb);

    const int t = threadIdx.x;
    const int lane = t & 31, w = t >> 5;
    const int g = lane >> 4, lm = lane & 15;
    const int wm = (w & 3) * 32;    // wave row offset within 128-tile
    const int wn = (w >> 2) * 64;   // wave col offset within 128-tile
    const int m0 = blockIdx.y * 128;
    const int n0 = blockIdx.x * 128;

    v8f c[2][4];
#pragma unroll
    for (int i = 0; i < 2; ++i)
#pragma unroll
        for (int j = 0; j < 4; ++j) c[i][j] = vzero8();

    for (int k0 = 0; k0 < 1024; k0 += 32) {
        __syncthreads();
        if (k0 + 32 < 1024)
            __builtin_prefetch(act + (size_t)(m0 + (t >> 3)) * 1024 + k0 + 32, 0, 1);
#pragma unroll
        for (int i = 0; i < 4; ++i) {                 // A tile: 128 rows x 32 f32 -> bf16
            int lin = t + i * 256;
            int row = lin >> 3, c4 = lin & 7;
            float4 d = *(const float4*)(act + (size_t)(m0 + row) * 1024 + k0 + c4 * 4);
            *(uint2*)(As + row * 40 + c4 * 4) = pack4bf(d);
        }
#pragma unroll
        for (int i = 0; i < 4; ++i) {                 // B tile: rows = out features f
            int lin = t + i * 256;
            int fr = lin >> 3, c4 = lin & 7;
            int f = n0 + fr;
            int hh = f >> 6, hs = f & 63;             // W[hs][h][d] contiguous in d
            float4 d = *(const float4*)(W + ((size_t)hs * 16 + hh) * 1024 + k0 + c4 * 4);
            *(uint2*)(Bs + fr * 40 + c4 * 4) = pack4bf(d);
        }
        __syncthreads();

        v16bf a[2], bfr[4];
#pragma unroll
        for (int mt = 0; mt < 2; ++mt)
            a[mt] = loadFragA(As + (wm + mt * 16 + lm) * 40 + g * 16);
#pragma unroll
        for (int nt = 0; nt < 4; ++nt)
            bfr[nt] = loadFragB(Bs + (wn + nt * 16 + lm) * 40 + g * 16);
#pragma unroll
        for (int mt = 0; mt < 2; ++mt)
#pragma unroll
            for (int nt = 0; nt < 4; ++nt)
                c[mt][nt] = bfwmma(a[mt], bfr[nt], c[mt][nt]);
    }

#pragma unroll
    for (int nt = 0; nt < 4; ++nt) {
        int f = n0 + wn + nt * 16 + lm;
        float bv_ = bias[f];
        int hh = f >> 6, hs = f & 63;
#pragma unroll
        for (int mt = 0; mt < 2; ++mt)
#pragma unroll
            for (int v = 0; v < 8; ++v) {
                int r = m0 + wm + mt * 16 + v + 8 * g;
                int bb = r >> 11, s = r & 2047;
                out[(((size_t)bb * 16 + hh) * 2048 + s) * 64 + hs] = f2bf(c[mt][nt][v] + bv_);
            }
    }
}

// ---------------------------------------------------------------------------
// Kernel 2: flash attention per (b,h).  WG = 256 q-rows, 8 waves x 32 rows,
// streaming 64-key blocks.  K tile is staged with async LDS-DMA (bf16,
// layout-preserving); V needs a transpose so it stays on the VGPR path.
// Online softmax init (m=0,l=1) models the always-attendable dummy column.
// ---------------------------------------------------------------------------
__global__ __launch_bounds__(256) void attn_kernel(
    const u16* __restrict__ qb, const u16* __restrict__ kb, const u16* __restrict__ vb,
    const int* __restrict__ amask, const float* __restrict__ pos, u16* __restrict__ z)
{
    __shared__ u16 Kt[64 * 72];            // [key][hs]  (B-operand image for Q.K^T)
    __shared__ u16 Vt[64 * 72];            // [hs][key]  (B-operand image for P.V)
    __shared__ float posm[64];             // bias+mask per key
    __shared__ u16 Pb[8 * 32 * 72];        // per-wave P round-trip buffer

    const int h = blockIdx.y, b = blockIdx.z;
    const int t = threadIdx.x, lane = t & 31, w = t >> 5;
    const int g = lane >> 4, lm = lane & 15;
    const size_t hoff = ((size_t)b * 16 + h) * 2048 * 64;
    const u16* Q = qb + hoff;
    const u16* K = kb + hoff;
    const u16* V = vb + hoff;
    const int q0 = blockIdx.x * 256 + w * 32;
    u16* Pw = Pb + w * 32 * 72;

    // Q fragments resident in registers for the whole key loop
    v16bf qf[2][2];
#pragma unroll
    for (int mt = 0; mt < 2; ++mt)
#pragma unroll
        for (int kk = 0; kk < 2; ++kk)
            qf[mt][kk] = loadFragA(Q + (size_t)(q0 + mt * 16 + lm) * 64 + kk * 32 + g * 16);

    v8f acc[2][4];
    v8f mrow[2], lrow[2];
#pragma unroll
    for (int mt = 0; mt < 2; ++mt) {
        mrow[mt] = vzero8();               // dummy column: score 0
        lrow[mt] = vzero8();
#pragma unroll
        for (int v = 0; v < 8; ++v) lrow[mt][v] = 1.f;  // exp(0-0)=1
#pragma unroll
        for (int ht = 0; ht < 4; ++ht) acc[mt][ht] = vzero8();
    }

    for (int j = 0; j < 2048; j += 64) {
        __syncthreads();
#pragma unroll
        for (int i = 0; i < 2; ++i) {      // K block 64x64 bf16 via async LDS-DMA
            int lin = t + i * 256;
            int row = lin >> 3, c8 = lin & 7;
            async_ld_b128(Kt + row * 72 + c8 * 8,
                          K + (size_t)(j + row) * 64 + c8 * 8);
        }
#pragma unroll
        for (int i = 0; i < 16; ++i) {     // V block, transposed into [hs][key]
            int lin = t + i * 256;
            int hs = lin & 63, kr = lin >> 6;
            Vt[hs * 72 + kr] = V[(size_t)(j + kr) * 64 + hs];
        }
        if (t < 64) {
            int key = j + t;
            float pv = pos[((size_t)b * 2048 + key) * 16 + h];
            posm[t] = amask[b * 2048 + key] ? pv : -1e30f;
        }
        wait_async0();                     // ASYNCcnt -> 0 for this wave's DMA
        __syncthreads();

        // S = Q.K^T  (K-depth 64 = two x32 WMMA steps)
        v8f s[2][4];
#pragma unroll
        for (int nt = 0; nt < 4; ++nt) {
            v16bf kf0 = loadFragB(Kt + (nt * 16 + lm) * 72 + 0 + g * 16);
            v16bf kf1 = loadFragB(Kt + (nt * 16 + lm) * 72 + 32 + g * 16);
#pragma unroll
            for (int mt = 0; mt < 2; ++mt) {
                v8f tmp = bfwmma(qf[mt][0], kf0, vzero8());
                s[mt][nt] = bfwmma(qf[mt][1], kf1, tmp);
            }
        }
        // scale 1/sqrt(64), add positional bias, apply mask (additive -1e30)
#pragma unroll
        for (int nt = 0; nt < 4; ++nt) {
            float pm = posm[nt * 16 + lm];
#pragma unroll
            for (int mt = 0; mt < 2; ++mt)
#pragma unroll
                for (int v = 0; v < 8; ++v)
                    s[mt][nt][v] = s[mt][nt][v] * 0.125f + pm;
        }
        // online softmax per q-row (rows live in 16-lane groups of C layout)
#pragma unroll
        for (int mt = 0; mt < 2; ++mt) {
            v8f t8;
#pragma unroll
            for (int v = 0; v < 8; ++v)
                t8[v] = fmaxf(fmaxf(s[mt][0][v], s[mt][1][v]),
                              fmaxf(s[mt][2][v], s[mt][3][v]));
#pragma unroll
            for (int off = 1; off <= 8; off <<= 1)
#pragma unroll
                for (int v = 0; v < 8; ++v)
                    t8[v] = fmaxf(t8[v], __shfl_xor(t8[v], off, 32));
#pragma unroll
            for (int v = 0; v < 8; ++v) {
                float mnew = fmaxf(mrow[mt][v], t8[v]);
                float sc = __expf(mrow[mt][v] - mnew);
                mrow[mt][v] = mnew;
                lrow[mt][v] *= sc;
#pragma unroll
                for (int ht = 0; ht < 4; ++ht) acc[mt][ht][v] *= sc;
            }
            v8f r8;
#pragma unroll
            for (int v = 0; v < 8; ++v) {
                float e0 = __expf(s[mt][0][v] - mrow[mt][v]);
                float e1 = __expf(s[mt][1][v] - mrow[mt][v]);
                float e2 = __expf(s[mt][2][v] - mrow[mt][v]);
                float e3 = __expf(s[mt][3][v] - mrow[mt][v]);
                s[mt][0][v] = e0; s[mt][1][v] = e1; s[mt][2][v] = e2; s[mt][3][v] = e3;
                r8[v] = (e0 + e1) + (e2 + e3);
            }
#pragma unroll
            for (int off = 1; off <= 8; off <<= 1)
#pragma unroll
                for (int v = 0; v < 8; ++v)
                    r8[v] += __shfl_xor(r8[v], off, 32);
#pragma unroll
            for (int v = 0; v < 8; ++v) lrow[mt][v] += r8[v];
            // C-layout -> A-layout round trip through per-wave LDS buffer
#pragma unroll
            for (int nt = 0; nt < 4; ++nt)
#pragma unroll
                for (int v = 0; v < 8; ++v)
                    Pw[(mt * 16 + v + 8 * g) * 72 + nt * 16 + lm] = f2bf(s[mt][nt][v]);
        }
        // acc += P.V  (K-depth 64 keys = two x32 steps)
#pragma unroll
        for (int k2 = 0; k2 < 2; ++k2) {
            v16bf vf[4];
#pragma unroll
            for (int ht = 0; ht < 4; ++ht)
                vf[ht] = loadFragB(Vt + (ht * 16 + lm) * 72 + k2 * 32 + g * 16);
#pragma unroll
            for (int mt = 0; mt < 2; ++mt) {
                v16bf pf = loadFragA(Pw + (mt * 16 + lm) * 72 + k2 * 32 + g * 16);
#pragma unroll
                for (int ht = 0; ht < 4; ++ht)
                    acc[mt][ht] = bfwmma(pf, vf[ht], acc[mt][ht]);
            }
        }
    }

    // normalize and store z[b*2048+q][h*64+hs] bf16
#pragma unroll
    for (int mt = 0; mt < 2; ++mt)
#pragma unroll
        for (int v = 0; v < 8; ++v) {
            float inv = 1.f / lrow[mt][v];
            int r = q0 + mt * 16 + v + 8 * g;
#pragma unroll
            for (int ht = 0; ht < 4; ++ht) {
                int col = h * 64 + ht * 16 + lm;
                z[((size_t)b * 2048 + r) * 1024 + col] = f2bf(acc[mt][ht][v] * inv);
            }
        }
}

// ---------------------------------------------------------------------------
// Kernel 3: output projection.  attn[4096x1024] = z(bf16) x W_O^T.
// A tile (already bf16) staged with async LDS-DMA; W_O rows converted f32->bf16.
// ---------------------------------------------------------------------------
__global__ __launch_bounds__(256) void oproj_kernel(
    const u16* __restrict__ z, const float* __restrict__ Wo, float* __restrict__ attn)
{
    __shared__ u16 As[128 * 40];
    __shared__ u16 Bs[128 * 40];

    const int t = threadIdx.x;
    const int lane = t & 31, w = t >> 5;
    const int g = lane >> 4, lm = lane & 15;
    const int wm = (w & 3) * 32;
    const int wn = (w >> 2) * 64;
    const int m0 = blockIdx.y * 128;
    const int n0 = blockIdx.x * 128;

    v8f c[2][4];
#pragma unroll
    for (int i = 0; i < 2; ++i)
#pragma unroll
        for (int j = 0; j < 4; ++j) c[i][j] = vzero8();

    for (int k0 = 0; k0 < 1024; k0 += 32) {
        __syncthreads();
        if (k0 + 32 < 1024)
            __builtin_prefetch(Wo + (size_t)(n0 + (t >> 3)) * 1024 + k0 + 32, 0, 1);
#pragma unroll
        for (int i = 0; i < 2; ++i) {      // A tile bf16: async LDS-DMA, 128 rows x 32
            int lin = t + i * 256;
            int row = lin >> 2, c8 = lin & 3;
            async_ld_b128(As + row * 40 + c8 * 8,
                          z + (size_t)(m0 + row) * 1024 + k0 + c8 * 8);
        }
#pragma unroll
        for (int i = 0; i < 4; ++i) {      // B tile from W_O rows (f32 -> bf16)
            int lin = t + i * 256;
            int er = lin >> 3, c4 = lin & 7;
            float4 d = *(const float4*)(Wo + (size_t)(n0 + er) * 1024 + k0 + c4 * 4);
            *(uint2*)(Bs + er * 40 + c4 * 4) = pack4bf(d);
        }
        wait_async0();
        __syncthreads();

        v16bf a[2], bfr[4];
#pragma unroll
        for (int mt = 0; mt < 2; ++mt)
            a[mt] = loadFragA(As + (wm + mt * 16 + lm) * 40 + g * 16);
#pragma unroll
        for (int nt = 0; nt < 4; ++nt)
            bfr[nt] = loadFragB(Bs + (wn + nt * 16 + lm) * 40 + g * 16);
#pragma unroll
        for (int mt = 0; mt < 2; ++mt)
#pragma unroll
            for (int nt = 0; nt < 4; ++nt)
                c[mt][nt] = bfwmma(a[mt], bfr[nt], c[mt][nt]);
    }

#pragma unroll
    for (int nt = 0; nt < 4; ++nt) {
        int f = n0 + wn + nt * 16 + lm;
#pragma unroll
        for (int mt = 0; mt < 2; ++mt)
#pragma unroll
            for (int v = 0; v < 8; ++v) {
                int r = m0 + wm + mt * 16 + v + 8 * g;
                attn[(size_t)r * 1024 + f] = c[mt][nt][v];
            }
    }
}

// ---------------------------------------------------------------------------
// Kernel 4: LayerNorm over 1024 features, one workgroup per row.
// ---------------------------------------------------------------------------
__global__ __launch_bounds__(256) void ln_kernel(
    const float* __restrict__ attn, const float* __restrict__ gamma,
    const float* __restrict__ beta, float* __restrict__ out)
{
    const int row = blockIdx.x;
    const float* x = attn + (size_t)row * 1024;
    const int t = threadIdx.x, lane = t & 31, w = t >> 5;

    float xs[4];
    float s = 0.f, s2 = 0.f;
#pragma unroll
    for (int i = 0; i < 4; ++i) {
        float v = x[t + i * 256];
        xs[i] = v; s += v; s2 += v * v;
    }
#pragma unroll
    for (int off = 16; off >= 1; off >>= 1) {
        s  += __shfl_xor(s,  off, 32);
        s2 += __shfl_xor(s2, off, 32);
    }
    __shared__ float ws1[8], ws2[8];
    if (lane == 0) { ws1[w] = s; ws2[w] = s2; }
    __syncthreads();
    float ts = 0.f, ts2 = 0.f;
#pragma unroll
    for (int i = 0; i < 8; ++i) { ts += ws1[i]; ts2 += ws2[i]; }
    float mu = ts * (1.f / 1024.f);
    float var = ts2 * (1.f / 1024.f) - mu * mu;
    float rs = rsqrtf(var + 1e-5f);
#pragma unroll
    for (int i = 0; i < 4; ++i) {
        int cidx = t + i * 256;
        out[(size_t)row * 1024 + cidx] = (xs[i] - mu) * rs * gamma[cidx] + beta[cidx];
    }
}

// ---------------------------------------------------------------------------
extern "C" void kernel_launch(void* const* d_in, const int* in_sizes, int n_in,
                              void* d_out, int out_size, void* d_ws, size_t ws_size,
                              hipStream_t stream)
{
    const float* query = (const float*)d_in[0];
    const float* key_  = (const float*)d_in[1];
    const float* value = (const float*)d_in[2];
    const int*   amask = (const int*)d_in[3];
    const float* pos   = (const float*)d_in[4];
    const float* Wq = (const float*)d_in[5];
    const float* bq = (const float*)d_in[6];
    const float* Wk = (const float*)d_in[7];
    const float* bk = (const float*)d_in[8];
    const float* Wv = (const float*)d_in[9];
    const float* bv = (const float*)d_in[10];
    const float* Wo = (const float*)d_in[11];
    const float* gamma = (const float*)d_in[12];
    const float* beta  = (const float*)d_in[13];
    float* out = (float*)d_out;

    char* ws = (char*)d_ws;
    u16*   qb   = (u16*)(ws);                                   // 8 MB  [B][H][S][64] bf16
    u16*   kb   = (u16*)(ws + (size_t)8  * 1024 * 1024);        // 8 MB
    u16*   vb   = (u16*)(ws + (size_t)16 * 1024 * 1024);        // 8 MB
    u16*   zb   = (u16*)(ws + (size_t)24 * 1024 * 1024);        // 8 MB  [4096][1024] bf16
    float* attn = (float*)(ws + (size_t)32 * 1024 * 1024);      // 16 MB [4096][1024] f32

    qkv_proj_kernel<<<dim3(8, 32, 3), 256, 0, stream>>>(
        query, key_, value, Wq, Wk, Wv, bq, bk, bv, qb, kb, vb);
    attn_kernel<<<dim3(8, 16, 2), 256, 0, stream>>>(qb, kb, vb, amask, pos, zb);
    oproj_kernel<<<dim3(8, 32), 256, 0, stream>>>(zb, Wo, attn);
    ln_kernel<<<dim3(4096), 256, 0, stream>>>(attn, gamma, beta, out);
}
